// ADFA_11931419148374
// MI455X (gfx1250) — matrix operand order repaired and checked by reading, not compiled, not run on hardware
//
#include <hip/hip_runtime.h>
#include <math.h>

// ---------------------------------------------------------------- constants
#define NB    2
#define HW    3136      // 56*56
#define DIMC  1792
#define SIDE  56

typedef unsigned short u16;
typedef __bf16 bf16x16 __attribute__((ext_vector_type(16)));
typedef float  f32x8   __attribute__((ext_vector_type(8)));

union FragAB { bf16x16 v; uint4 q[2]; };

__device__ __forceinline__ u16 f2bf(float f) {
    unsigned u = __float_as_uint(f);
    unsigned r = 0x7FFFu + ((u >> 16) & 1u);
    return (u16)((u + r) >> 16);
}

// CDNA5 async global->LDS copy (ASYNCcnt-tracked DMA, no VGPR round trip).
// ldsAddr: 32-bit LDS byte address (low 32 bits of generic shared pointer),
// gptr:    64-bit global address.
__device__ __forceinline__ void async_copy16(unsigned ldsAddr, const void* gptr) {
    asm volatile("global_load_async_to_lds_b128 %0, %1, off"
                 :: "v"(ldsAddr), "v"(gptr) : "memory");
}
__device__ __forceinline__ void async_wait0() {
    asm volatile("s_wait_asynccnt 0" ::: "memory");
}

// ---------------------------------------------------------------- reductions
__device__ __forceinline__ float bsum(float v, float* sm) {
    int tid = threadIdx.x;
    sm[tid] = v; __syncthreads();
    for (int o = 128; o > 0; o >>= 1) { if (tid < o) sm[tid] += sm[tid + o]; __syncthreads(); }
    float r = sm[0]; __syncthreads();
    return r;
}
__device__ __forceinline__ float bmax(float v, float* sm) {
    int tid = threadIdx.x;
    sm[tid] = v; __syncthreads();
    for (int o = 128; o > 0; o >>= 1) { if (tid < o) sm[tid] = fmaxf(sm[tid], sm[tid + o]); __syncthreads(); }
    float r = sm[0]; __syncthreads();
    return r;
}

// ---------------------------------------------------------------- init
__global__ void init_kernel(unsigned* cmaxbits) {
    if (threadIdx.x == 0 && blockIdx.x == 0) *cmaxbits = 0u;
}

// ---------------------------------------------------------------- 3x3 avg pool (pad 1, /9)
__global__ void pool3x3_kernel(const float* __restrict__ in, float* __restrict__ out,
                               int C, int H, int W) {
    int idx = blockIdx.x * 256 + threadIdx.x;
    int total = NB * C * H * W;
    if (idx >= total) return;
    int w = idx % W; int t = idx / W;
    int h = t % H;   t /= H;                    // t = b*C + c
    const float* src = in + (size_t)t * H * W;
    float s = 0.f;
    for (int dh = -1; dh <= 1; ++dh) {
        int hh = h + dh; if (hh < 0 || hh >= H) continue;
        for (int dw = -1; dw <= 1; ++dw) {
            int ww = w + dw; if (ww < 0 || ww >= W) continue;
            s += src[hh * W + ww];
        }
    }
    out[idx] = s * (1.0f / 9.0f);
}

// ---------------------------------------------------------------- half-pixel bilinear sample
__device__ __forceinline__ float bilinear(const float* __restrict__ img, int H, int W,
                                          float fy, float fx) {
    float y0f = floorf(fy), x0f = floorf(fx);
    float ay = fy - y0f, ax = fx - x0f;
    int y0 = (int)y0f, x0 = (int)x0f;
    int y1 = y0 + 1, x1 = x0 + 1;
    y0 = min(max(y0, 0), H - 1); y1 = min(max(y1, 0), H - 1);
    x0 = min(max(x0, 0), W - 1); x1 = min(max(x1, 0), W - 1);
    float v00 = img[y0 * W + x0], v01 = img[y0 * W + x1];
    float v10 = img[y1 * W + x0], v11 = img[y1 * W + x1];
    float top = v00 + ax * (v01 - v00);
    float bot = v10 + ax * (v11 - v10);
    return top + ay * (bot - top);
}

// assemble Z[b][p][c] (bf16) : channels 0-255 pool1, 256-767 upsampled pool2, 768-1791 pool3
__global__ void assembleZ_kernel(const float* __restrict__ q1, const float* __restrict__ q2,
                                 const float* __restrict__ q3, u16* __restrict__ Zb) {
    int row = blockIdx.x;                       // b*HW + p
    int c = blockIdx.y * 256 + threadIdx.x;
    if (c >= DIMC) return;
    int b = row / HW, p = row % HW;
    int h = p / SIDE, w = p % SIDE;
    float val;
    if (c < 256) {
        val = q1[(((size_t)b * 256 + c) * SIDE + h) * SIDE + w];
    } else if (c < 768) {
        const float* img = q2 + ((size_t)b * 512 + (c - 256)) * 784;
        val = bilinear(img, 28, 28, (h + 0.5f) * 0.5f - 0.5f, (w + 0.5f) * 0.5f - 0.5f);
    } else {
        const float* img = q3 + ((size_t)b * 1024 + (c - 768)) * 196;
        val = bilinear(img, 14, 14, (h + 0.5f) * 0.25f - 0.5f, (w + 0.5f) * 0.25f - 0.5f);
    }
    Zb[(size_t)row * DIMC + c] = f2bf(val);
}

// ---------------------------------------------------------------- weight prep
__global__ void convW_kernel(const float* __restrict__ W, u16* __restrict__ Wbf) {
    int idx = blockIdx.x * 256 + threadIdx.x;
    if (idx >= DIMC * DIMC) return;
    int o = idx / DIMC, c = idx % DIMC;
    Wbf[idx] = f2bf(W[(size_t)o * 1794 + c]);
}
__global__ void wxy_kernel(const float* __restrict__ W, float* __restrict__ wx,
                           float* __restrict__ wy) {
    int o = blockIdx.x * 256 + threadIdx.x;
    if (o >= DIMC) return;
    wx[o] = W[(size_t)o * 1794 + 1792];
    wy[o] = W[(size_t)o * 1794 + 1793];
}

// ---------------------------------------------------------------- C transpose + per-row stats
__global__ void transC_kernel(const float* __restrict__ C, float* __restrict__ Ctf,
                              u16* __restrict__ Ctbf) {
    int p = blockIdx.x;
    int c = blockIdx.y * 256 + threadIdx.x;
    if (c >= DIMC) return;
    float v = C[(size_t)c * HW + p];
    Ctf [(size_t)p * DIMC + c] = v;
    Ctbf[(size_t)p * DIMC + c] = f2bf(v);
}

// per row p of C^T: softmax stats, sum t*log t, sum of squares (cent2)
__global__ __launch_bounds__(256) void cstats_kernel(const float* __restrict__ Ctf,
        float* __restrict__ tm, float* __restrict__ tSinv,
        float* __restrict__ negent, float* __restrict__ cent2) {
    __shared__ float sm[256];
    int p = blockIdx.x, tid = threadIdx.x;
    const float* rowp = Ctf + (size_t)p * DIMC;
    float x[7]; float mx = -1e30f, sq = 0.f;
#pragma unroll
    for (int i = 0; i < 7; ++i) { x[i] = rowp[tid + i * 256]; mx = fmaxf(mx, x[i]); sq += x[i] * x[i]; }
    mx = bmax(mx, sm);
    float se = 0.f, sex = 0.f;
#pragma unroll
    for (int i = 0; i < 7; ++i) { float e = __expf(x[i] - mx); se += e; sex += e * x[i]; }
    se  = bsum(se, sm);
    sex = bsum(sex, sm);
    sq  = bsum(sq, sm);
    if (tid == 0) {
        tm[p] = mx; tSinv[p] = 1.f / se;
        negent[p] = sex / se - mx - logf(se);   // sum t*log t
        cent2[p] = sq;
    }
}

// ---------------------------------------------------------------- WMMA GEMM (bf16 x bf16 -> f32)
// Double-buffered LDS, filled by CDNA5 async global->LDS DMA.
// A[M][K] row-major bf16 (batched by z), Bt[N][K] row-major bf16 (shared across batch).
// MODE 1: out = phi[b][p][o] = acc + bias[o] + wx[o]*xx(p) + wy[o]*yy(p)
// MODE 2: out = dist[b][p][q] = (feat2[b][p] + cent2[q] - 2*acc) * kld[b]; track global max
template <int MODE>
__global__ __launch_bounds__(256) void wmma_gemm_kernel(
        const u16* __restrict__ A, const u16* __restrict__ Bt,
        float* __restrict__ out,
        const float* __restrict__ e0, const float* __restrict__ e1,
        const float* __restrict__ e2,
        const float* __restrict__ kld, unsigned* __restrict__ cmaxbits) {
    const int K = DIMC;
    const int z = blockIdx.z;
    const u16* Ab = A + (size_t)z * HW * K;
    const int m0 = blockIdx.y * 64, n0 = blockIdx.x * 64;

    // [2 buffers][64 rows][32 cols + 8 pad]
    __shared__ __align__(16) u16 As[2][64 * 40];
    __shared__ __align__(16) u16 Bs[2][64 * 40];
    const unsigned BUFB = (unsigned)(64 * 40 * sizeof(u16));   // 5120 bytes

    const int tid = threadIdx.x;
    const int lane = tid & 31, wid = tid >> 5;
    const int wm = wid & 1;        // 2 waves over M (32 rows each)
    const int wn = wid >> 1;       // 4 waves over N (16 cols each)

    const int lrow = tid >> 2;                 // 0..63
    const int lc   = (tid & 3) * 8;            // 0,8,16,24 elements (16B chunks)
    const u16* gA = Ab + (size_t)(m0 + lrow) * K + lc;
    const u16* gB = Bt + (size_t)(n0 + lrow) * K + lc;
    const unsigned ldsA = (unsigned)(uintptr_t)&As[0][lrow * 40 + lc];
    const unsigned ldsB = (unsigned)(uintptr_t)&Bs[0][lrow * 40 + lc];

    f32x8 acc0 = {0.f,0.f,0.f,0.f,0.f,0.f,0.f,0.f};
    f32x8 acc1 = {0.f,0.f,0.f,0.f,0.f,0.f,0.f,0.f};

    const int halfA = (lane < 16) ? 0 : 8;
    const int rowA  = wm * 32 + (lane & 15);
    const int rowB  = wn * 16 + (lane & 15);
    const int halfB = (lane < 16) ? 0 : 8;

    // prologue: DMA tile 0 into buffer 0
    async_copy16(ldsA, gA);
    async_copy16(ldsB, gB);
    async_wait0();
    __syncthreads();

    const int NK = K / 32;                      // 56
    for (int i = 0; i < NK; ++i) {
        const int cur = i & 1;
        if (i + 1 < NK) {                       // DMA next tile into other buffer
            const unsigned nb = (unsigned)((i + 1) & 1) * BUFB;
            async_copy16(ldsA + nb, gA + (i + 1) * 32);
            async_copy16(ldsB + nb, gB + (i + 1) * 32);
        }

        const u16* asb = As[cur];
        const u16* bsb = Bs[cur];
        FragAB a0, a1, b;
        a0.q[0] = *(const uint4*)&asb[(rowA     ) * 40 + halfA];
        a0.q[1] = *(const uint4*)&asb[(rowA     ) * 40 + halfA + 16];
        a1.q[0] = *(const uint4*)&asb[(rowA + 16) * 40 + halfA];
        a1.q[1] = *(const uint4*)&asb[(rowA + 16) * 40 + halfA + 16];
        b.q[0]  = *(const uint4*)&bsb[rowB * 40 + halfB];
        b.q[1]  = *(const uint4*)&bsb[rowB * 40 + halfB + 16];

        acc0 = __builtin_amdgcn_wmma_f32_16x16x32_bf16(false, a0.v, false, b.v,
                                                       (short)0, acc0, false, false);
        acc1 = __builtin_amdgcn_wmma_f32_16x16x32_bf16(false, a1.v, false, b.v,
                                                       (short)0, acc1, false, false);

        async_wait0();      // prefetch landed in LDS
        __syncthreads();    // all waves done reading cur / writing next
    }

    const int ng = n0 + wn * 16 + (lane & 15);
    float cmaxloc = 0.f;
    float bias = 0.f, wxv = 0.f, wyv = 0.f, c2 = 0.f, kv = 0.f;
    if (MODE == 1) { bias = e0[ng]; wxv = e1[ng]; wyv = e2[ng]; }
    else           { c2 = e1[ng];   kv = kld[z]; }

    float* outb = (MODE == 1) ? out + (size_t)z * HW * DIMC
                              : out + (size_t)z * HW * HW;

#pragma unroll
    for (int sub = 0; sub < 2; ++sub) {
        f32x8 acc = sub ? acc1 : acc0;
        int mb = m0 + wm * 32 + sub * 16 + ((lane < 16) ? 0 : 8);
#pragma unroll
        for (int r = 0; r < 8; ++r) {
            int mg = mb + r;
            if (MODE == 1) {
                float xx = -1.f + (2.f / 55.f) * (float)(mg % SIDE);
                float yy = -1.f + (2.f / 55.f) * (float)(mg / SIDE);
                outb[(size_t)mg * DIMC + ng] = acc[r] + bias + wxv * xx + wyv * yy;
            } else {
                float f2 = e0[(size_t)z * HW + mg];
                float v = (f2 + c2 - 2.f * acc[r]) * kv;
                outb[(size_t)mg * HW + ng] = v;
                float a = v * v, bq = (v - 1.f) * (v - 1.f);
                cmaxloc = fmaxf(cmaxloc, fmaxf(a, bq));
            }
        }
    }

    if (MODE == 2) {
        __syncthreads();
        float* red = (float*)&As[0][0];
        red[tid] = cmaxloc; __syncthreads();
        for (int o = 128; o > 0; o >>= 1) {
            if (tid < o) red[tid] = fmaxf(red[tid], red[tid + o]);
            __syncthreads();
        }
        if (tid == 0) atomicMax(cmaxbits, __float_as_uint(red[0]));
    }
}

// ---------------------------------------------------------------- ECA
__global__ void colmean_kernel(const float* __restrict__ phi, float* __restrict__ y) {
    int o = blockIdx.x * 256 + threadIdx.x;
    int b = blockIdx.y;
    if (o >= DIMC) return;
    const float* base = phi + (size_t)b * HW * DIMC + o;
    float s = 0.f;
    for (int p = 0; p < HW; ++p) s += base[(size_t)p * DIMC];
    y[b * DIMC + o] = s * (1.f / (float)HW);
}

__global__ void eca_kernel(const float* __restrict__ y, const float* __restrict__ we,
                           float* __restrict__ gmul) {
    int idx = blockIdx.x * 256 + threadIdx.x;
    if (idx >= NB * DIMC) return;
    int b = idx / DIMC, o = idx % DIMC;
    float s = 0.f;
#pragma unroll
    for (int j = 0; j < 9; ++j) {
        int oo = o + j - 4;
        if (oo >= 0 && oo < DIMC) s += we[j] * y[b * DIMC + oo];
    }
    float g = 1.f / (1.f + __expf(-s));
    gmul[idx] = 1.f + 0.1f * g;
}

// apply gate, emit bf16 phi, and per-(b,p) row stats: feat2, logsumexp, sum_t*phi
__global__ __launch_bounds__(256) void gatestats_kernel(
        const float* __restrict__ phi, const float* __restrict__ gmul,
        const float* __restrict__ Ctf, const float* __restrict__ tm,
        const float* __restrict__ tSinv,
        u16* __restrict__ phibf, float* __restrict__ feat2,
        float* __restrict__ lse, float* __restrict__ tphi) {
    __shared__ float sm[256];
    int row = blockIdx.x, tid = threadIdx.x;
    int b = row / HW, p = row % HW;
    const float* src = phi + (size_t)row * DIMC;
    const float* gm  = gmul + b * DIMC;
    const float* ct  = Ctf + (size_t)p * DIMC;
    float v[7]; float mx = -1e30f, sq = 0.f;
#pragma unroll
    for (int i = 0; i < 7; ++i) {
        int o = tid + i * 256;
        float x = src[o] * gm[o];
        v[i] = x; mx = fmaxf(mx, x); sq += x * x;
        phibf[(size_t)row * DIMC + o] = f2bf(x);
    }
    mx = bmax(mx, sm);
    float se = 0.f, ts = 0.f;
    float tmv = tm[p], tsv = tSinv[p];
#pragma unroll
    for (int i = 0; i < 7; ++i) {
        int o = tid + i * 256;
        se += __expf(v[i] - mx);
        ts += __expf(ct[o] - tmv) * tsv * v[i];
    }
    se = bsum(se, sm); sq = bsum(sq, sm); ts = bsum(ts, sm);
    if (tid == 0) { feat2[row] = sq; lse[row] = mx + logf(se); tphi[row] = ts; }
}

// kldist[b] = sum_p ( negent[p] - tphi[b][p] + lse[b][p] )
__global__ __launch_bounds__(256) void kld_kernel(const float* __restrict__ negent,
        const float* __restrict__ tphi, const float* __restrict__ lse,
        float* __restrict__ kld) {
    __shared__ float sm[256];
    int b = blockIdx.x, tid = threadIdx.x;
    float s = 0.f;
    for (int p = tid; p < HW; p += 256)
        s += negent[p] - tphi[b * HW + p] + lse[b * HW + p];
    s = bsum(s, sm);
    if (tid == 0) kld[b] = s;
}

// ---------------------------------------------------------------- Sinkhorn soft top-k per row
__global__ __launch_bounds__(256) void sinkhorn_kernel(const float* __restrict__ dist,
        const unsigned* __restrict__ cmaxbits, float* __restrict__ out) {
    __shared__ float r0[256], r1[256];
    const int row = blockIdx.x, tid = threadIdx.x;
    const float* s = dist + (size_t)row * HW;
    const float Cmax = __uint_as_float(*cmaxbits);
    const float inv = 10.f / Cmax;               // 1/(Cmax*EPS), EPS=0.1
    float sv[13], G0[13], G1[13], U[13];
#pragma unroll
    for (int i = 0; i < 13; ++i) {
        int idx = tid + i * 256;
        if (idx < HW) {
            float x = s[idx];
            sv[i] = x;
            G0[i] = __expf(-x * x * inv);
            G1[i] = __expf(-(x - 1.f) * (x - 1.f) * inv);
        } else { sv[i] = 0.f; G0[i] = 0.f; G1[i] = 0.f; }
        U[i] = 0.f;
    }
    float v0 = 0.5f, v1 = 0.5f;
    const float mu = 1.f / (float)HW;
    const float nu0 = 3.f / (float)HW;
    const float nu1 = (float)(HW - 3) / (float)HW;

    for (int it = 0; it < 200; ++it) {
        float l0 = 0.f, l1 = 0.f;
#pragma unroll
        for (int i = 0; i < 13; ++i) {
            float den = G0[i] * v0 + G1[i] * v1;
            float u = (den > 0.f) ? (mu / den) : 0.f;
            U[i] = u;
            l0 += u * G0[i];
            l1 += u * G1[i];
        }
        r0[tid] = l0; r1[tid] = l1; __syncthreads();
        for (int o = 128; o > 0; o >>= 1) {
            if (tid < o) { r0[tid] += r0[tid + o]; r1[tid] += r1[tid + o]; }
            __syncthreads();
        }
        v0 = nu0 / r0[0];
        v1 = nu1 / r1[0];
        __syncthreads();
    }

    float dl = 0.f;
#pragma unroll
    for (int i = 0; i < 13; ++i)
        dl += (float)HW * U[i] * G0[i] * v0 * sv[i];
    r0[tid] = dl; __syncthreads();
    for (int o = 128; o > 0; o >>= 1) {
        if (tid < o) r0[tid] += r0[tid + o];
        __syncthreads();
    }
    if (tid == 0) out[row] = sqrtf(r0[0]);
}

// ---------------------------------------------------------------- launcher
extern "C" void kernel_launch(void* const* d_in, const int* in_sizes, int n_in,
                              void* d_out, int out_size, void* d_ws, size_t ws_size,
                              hipStream_t stream) {
    const float* p1  = (const float*)d_in[0];   // [2,256,56,56]
    const float* p2  = (const float*)d_in[1];   // [2,512,28,28]
    const float* p3  = (const float*)d_in[2];   // [2,1024,14,14]
    const float* wco = (const float*)d_in[3];   // [1792,1794]
    const float* bco = (const float*)d_in[4];   // [1792]
    const float* wec = (const float*)d_in[5];   // [9]
    const float* Cc  = (const float*)d_in[6];   // [1792,3136]
    float* out = (float*)d_out;

    char* ws = (char*)d_ws;
    auto al = [](size_t x) { return (x + 255) & ~(size_t)255; };
    size_t off = 0;

    // ---- persistent region
    unsigned* cmaxb = (unsigned*)(ws + off); off = al(off + 16);
    u16* phibf = (u16*)(ws + off);  off = al(off + (size_t)NB * HW * DIMC * 2);
    u16* ctbf  = (u16*)(ws + off);  off = al(off + (size_t)HW * DIMC * 2);
    float* tm     = (float*)(ws + off); off = al(off + (size_t)HW * 4);
    float* tSinv  = (float*)(ws + off); off = al(off + (size_t)HW * 4);
    float* negent = (float*)(ws + off); off = al(off + (size_t)HW * 4);
    float* cent2  = (float*)(ws + off); off = al(off + (size_t)HW * 4);
    float* feat2  = (float*)(ws + off); off = al(off + (size_t)NB * HW * 4);
    float* lse    = (float*)(ws + off); off = al(off + (size_t)NB * HW * 4);
    float* tphi   = (float*)(ws + off); off = al(off + (size_t)NB * HW * 4);
    float* ymean  = (float*)(ws + off); off = al(off + (size_t)NB * DIMC * 4);
    float* gmul   = (float*)(ws + off); off = al(off + (size_t)NB * DIMC * 4);
    float* wx     = (float*)(ws + off); off = al(off + (size_t)DIMC * 4);
    float* wy     = (float*)(ws + off); off = al(off + (size_t)DIMC * 4);
    float* kld    = (float*)(ws + off); off = al(off + (size_t)NB * 4);

    // ---- reusable region (dead before dist is written; dist aliases it)
    size_t reuse0 = off;
    float* pooled1 = (float*)(ws + off); off = al(off + (size_t)NB * 256  * HW  * 4);
    float* pooled2 = (float*)(ws + off); off = al(off + (size_t)NB * 512  * 784 * 4);
    float* pooled3 = (float*)(ws + off); off = al(off + (size_t)NB * 1024 * 196 * 4);
    u16*   Zb      = (u16*)  (ws + off); off = al(off + (size_t)NB * HW * DIMC * 2);
    u16*   Wbf     = (u16*)  (ws + off); off = al(off + (size_t)DIMC * DIMC * 2);
    float* phi     = (float*)(ws + off); off = al(off + (size_t)NB * HW * DIMC * 4);
    float* Ctf     = (float*)(ws + off); off = al(off + (size_t)HW * DIMC * 4);
    float* dist    = (float*)(ws + reuse0);     // 2*3136*3136*4 = 78.7MB <= reuse region

    init_kernel<<<1, 32, 0, stream>>>(cmaxb);

    pool3x3_kernel<<<(NB * 256  * HW  + 255) / 256, 256, 0, stream>>>(p1, pooled1, 256, 56, 56);
    pool3x3_kernel<<<(NB * 512  * 784 + 255) / 256, 256, 0, stream>>>(p2, pooled2, 512, 28, 28);
    pool3x3_kernel<<<(NB * 1024 * 196 + 255) / 256, 256, 0, stream>>>(p3, pooled3, 1024, 14, 14);

    transC_kernel<<<dim3(HW, 7), 256, 0, stream>>>(Cc, Ctf, ctbf);
    cstats_kernel<<<HW, 256, 0, stream>>>(Ctf, tm, tSinv, negent, cent2);

    convW_kernel<<<(DIMC * DIMC + 255) / 256, 256, 0, stream>>>(wco, Wbf);
    wxy_kernel<<<7, 256, 0, stream>>>(wco, wx, wy);

    assembleZ_kernel<<<dim3(NB * HW, 7), 256, 0, stream>>>(pooled1, pooled2, pooled3, Zb);

    // GEMM 1: phi[b][p][o] = Z @ W^T + bias + coord terms
    wmma_gemm_kernel<1><<<dim3(DIMC / 64, HW / 64, NB), 256, 0, stream>>>(
        Zb, Wbf, phi, bco, wx, wy, nullptr, nullptr);

    colmean_kernel<<<dim3(7, NB), 256, 0, stream>>>(phi, ymean);
    eca_kernel<<<(NB * DIMC + 255) / 256, 256, 0, stream>>>(ymean, wec, gmul);

    gatestats_kernel<<<NB * HW, 256, 0, stream>>>(phi, gmul, Ctf, tm, tSinv,
                                                  phibf, feat2, lse, tphi);
    kld_kernel<<<NB, 256, 0, stream>>>(negent, tphi, lse, kld);

    // GEMM 2: dist[b][p][q] = (feat2 + cent2 - 2*phi@C) * kld ; track global Ccost max
    wmma_gemm_kernel<2><<<dim3(HW / 64, HW / 64, NB), 256, 0, stream>>>(
        phibf, ctbf, dist, feat2, cent2, nullptr, kld, cmaxb);

    sinkhorn_kernel<<<NB * HW, 256, 0, stream>>>(dist, cmaxb, out);
}
